// TransformerCrossEncoderLayer_4715874091071
// MI455X (gfx1250) — compile-verified, matrix operand order checked by reading, not run on hardware
//
#include <hip/hip_runtime.h>
#include <hip/hip_bf16.h>
#include <math.h>

// ---------------------------------------------------------------------------
// Problem constants (from reference setup_inputs)
// ---------------------------------------------------------------------------
#define NN   512          // src seq len
#define MM   512          // tgt seq len
#define BB   2            // batch
#define CC   256          // channels
#define HH   8            // heads
#define DD   32           // head dim
#define FF   1024         // ffn dim
#define TOK  (NN * BB * CC)   // 262144 floats per token tensor
#define INV_SQRT_D 0.17677669529663687f   // 1/sqrt(32)

typedef __bf16 bf16_t;
typedef __attribute__((ext_vector_type(16))) __bf16 v16bf;
typedef __attribute__((ext_vector_type(8)))  float  v8f;

// ---------------------------------------------------------------------------
// WMMA fragment loaders (fp32 memory -> bf16 fragment)
// 16-bit A 16x32 layout: lanes 0-15 hold M=0..15; half = lane>>4 selects K-bank:
//   elems 0..7  : K = half*8 + 0..7
//   elems 8..15 : K = 16 + half*8 + 0..7
// B (K x N), transpose-symmetric mapping: lane&15 = N column, same K mapping.
// ---------------------------------------------------------------------------
__device__ __forceinline__ v16bf frag_row(const float* __restrict__ base, int ld,
                                          int row0, int col0, int lane) {
  const float* p = base + (long)(row0 + (lane & 15)) * ld + col0 + ((lane >> 4) & 1) * 8;
  v16bf f;
#pragma unroll
  for (int i = 0; i < 8; ++i) f[i] = (__bf16)p[i];
#pragma unroll
  for (int i = 0; i < 8; ++i) f[8 + i] = (__bf16)p[16 + i];
  return f;
}

// non-transposed B: B[k][n] row-major with leading dim ld; lane&15 = column n
__device__ __forceinline__ v16bf frag_colk(const float* __restrict__ base, int ld,
                                           int k0, int n0, int lane) {
  const float* p = base + (long)(k0 + ((lane >> 4) & 1) * 8) * ld + n0 + (lane & 15);
  v16bf f;
#pragma unroll
  for (int i = 0; i < 8; ++i) f[i] = (__bf16)p[(long)i * ld];
#pragma unroll
  for (int i = 0; i < 8; ++i) f[8 + i] = (__bf16)p[(long)(16 + i) * ld];
  return f;
}

// ---------------------------------------------------------------------------
// Generic batched WMMA GEMM:
//   Y[m][n] = alpha * sum_k A[m][k] * Bop[k][n] (+bias[n]) (+ReLU) (+resid)
//   BT=true : Bop[k][n] = Bm[n][k]  (PyTorch W row-major, X@W.T)
//   BT=false: Bop[k][n] = Bm[k][n]
// Batch bi = blockIdx.y; per-operand offset = (bi%Hb)*s?h + (bi/Hb)*s?b.
// Residual shares Y geometry. Block = 256 threads = 8 waves, 1 16x16 tile/wave.
// ---------------------------------------------------------------------------
template <bool BT>
__global__ __launch_bounds__(256) void gemm_wmma_bf16(
    const float* __restrict__ A, int lda, long sAh, long sAb,
    const float* __restrict__ Bm, int ldb, long sBh, long sBb,
    const float* __restrict__ bias, const float* __restrict__ resid,
    float* __restrict__ Y, int ldy, long sYh, long sYb,
    int Mr, int Nc, int Kc, int Hb, float alpha, int relu) {
  const int wave = threadIdx.x >> 5;
  const int lane = threadIdx.x & 31;
  const int bi = blockIdx.y;
  const long offA = (long)(bi % Hb) * sAh + (long)(bi / Hb) * sAb;
  const long offB = (long)(bi % Hb) * sBh + (long)(bi / Hb) * sBb;
  const long offY = (long)(bi % Hb) * sYh + (long)(bi / Hb) * sYb;

  const int tm = Mr >> 4, tn = Nc >> 4;
  const int tile = blockIdx.x * 8 + wave;
  if (tile >= tm * tn) return;
  const int m0 = (tile % tm) << 4;
  const int n0 = (tile / tm) << 4;

  v8f acc = {0.f, 0.f, 0.f, 0.f, 0.f, 0.f, 0.f, 0.f};
  const float* Ab = A + offA;
  const float* Bb = Bm + offB;
  for (int kk = 0; kk < Kc; kk += 32) {
    v16bf af = frag_row(Ab, lda, m0, kk, lane);
    v16bf bf = BT ? frag_row(Bb, ldb, n0, kk, lane)
                  : frag_colk(Bb, ldb, kk, n0, lane);
    acc = __builtin_amdgcn_wmma_f32_16x16x32_bf16(false, af, false, bf,
                                                  (short)0, acc, false, false);
  }
  const int nc = n0 + (lane & 15);
  const int half = (lane >> 4) & 1;
  const float bv = bias ? bias[nc] : 0.f;
#pragma unroll
  for (int r = 0; r < 8; ++r) {
    const int mr = m0 + r + half * 8;
    float y = alpha * acc[r] + bv;
    if (relu) y = fmaxf(y, 0.f);
    const long idx = (long)mr * ldy + nc + offY;
    if (resid) y += resid[idx];
    Y[idx] = y;
  }
}

// ---------------------------------------------------------------------------
// LayerNorm over C=256, one row (token) per block of 256 threads.
// Writes out_ln; if pos != null also writes out_lp = ln + pos.
// ---------------------------------------------------------------------------
__global__ __launch_bounds__(256) void ln_kernel(
    const float* __restrict__ X, const float* __restrict__ g,
    const float* __restrict__ bvec, const float* __restrict__ pos,
    float* __restrict__ out_ln, float* __restrict__ out_lp) {
  const long base = (long)blockIdx.x * CC;
  const int t = threadIdx.x, lane = t & 31, wave = t >> 5;
  __shared__ float red[8];
  const float x = X[base + t];

  float s = x;
#pragma unroll
  for (int o = 16; o; o >>= 1) s += __shfl_xor(s, o, 32);
  if (!lane) red[wave] = s;
  __syncthreads();
  float mean = 0.f;
#pragma unroll
  for (int i = 0; i < 8; ++i) mean += red[i];
  mean *= (1.f / CC);
  __syncthreads();

  const float dx = x - mean;
  s = dx * dx;
#pragma unroll
  for (int o = 16; o; o >>= 1) s += __shfl_xor(s, o, 32);
  if (!lane) red[wave] = s;
  __syncthreads();
  float var = 0.f;
#pragma unroll
  for (int i = 0; i < 8; ++i) var += red[i];
  var *= (1.f / CC);

  const float y = dx * rsqrtf(var + 1e-5f) * g[t] + bvec[t];
  out_ln[base + t] = y;
  if (out_lp) out_lp[base + t] = y + pos[base + t];
}

// ---------------------------------------------------------------------------
// u precompute: u[r,h,c] = sum_d q[r, h*32+d] * Wg[h*32+d, c]   (r = n*B+b)
// gb[r,h]     = sum_d q[r, h*32+d] * bg[h*32+d]
// ---------------------------------------------------------------------------
__global__ __launch_bounds__(256) void u_kernel(
    const float* __restrict__ Q, const float* __restrict__ Wg,
    const float* __restrict__ bg, float* __restrict__ U,
    float* __restrict__ GB) {
  const int r = blockIdx.x, t = threadIdx.x;
  __shared__ float qs[CC];
  qs[t] = Q[(long)r * CC + t];
  __syncthreads();
#pragma unroll
  for (int h = 0; h < HH; ++h) {
    float acc = 0.f;
#pragma unroll
    for (int d = 0; d < DD; ++d)
      acc += qs[h * DD + d] * Wg[(long)(h * DD + d) * CC + t];
    U[((long)r * HH + h) * CC + t] = acc;
  }
  if (t < HH) {
    float gb = 0.f;
#pragma unroll
    for (int d = 0; d < DD; ++d) gb += qs[t * DD + d] * bg[t * DD + d];
    GB[r * HH + t] = gb;
  }
}

// ---------------------------------------------------------------------------
// Fused RPE attention: one block per (b,n) token (r = n*B+b), 8 waves.
//  1. scores_g = G(512x256) @ u.T via WMMA (streams src_global once)
//  2. sc = Sqk + (scores_g + q.bg) / sqrt(d)
//  3. softmax over m per head (wave w <-> head w)
//  4. out = A @ V, + residual -> cur
// ---------------------------------------------------------------------------
__global__ __launch_bounds__(256) void rpe_attn_kernel(
    const float* __restrict__ Gin,  // (B,N,M,C)
    const float* __restrict__ U,    // (r,H,C)
    const float* __restrict__ GB,   // (r,H)
    const float* __restrict__ Sqk,  // (B,H,N,M), already scaled by 1/sqrt(d)
    const float* __restrict__ V,    // (M,B,C)
    const float* __restrict__ resid,
    float* __restrict__ out) {
  const int r = blockIdx.x;
  const int b = r % BB, n = r / BB;
  const int t = threadIdx.x, wave = t >> 5, lane = t & 31;
  const int hn = lane & 15, half = (lane >> 4) & 1;
  __shared__ float sc[HH * MM];  // 16 KB

  // Precompute B fragments (u.T: K=c, N=h, cols 8..15 zero) for all 8 k-steps.
  v16bf bfrag[8];
  const float* ub = U + (long)r * HH * CC;
#pragma unroll
  for (int ks = 0; ks < 8; ++ks) {
    v16bf f;
    if (hn < HH) {
      const float* p = ub + hn * CC + ks * 32 + half * 8;
#pragma unroll
      for (int i = 0; i < 8; ++i) f[i] = (__bf16)p[i];
#pragma unroll
      for (int i = 0; i < 8; ++i) f[8 + i] = (__bf16)p[16 + i];
    } else {
#pragma unroll
      for (int i = 0; i < 16; ++i) f[i] = (__bf16)0.f;
    }
    bfrag[ks] = f;
  }
  const float gb = (hn < HH) ? GB[r * HH + hn] : 0.f;
  const float* gbase = Gin + ((long)(b * NN + n)) * MM * CC;

  // 32 m-tiles of 16 rows; each wave handles 4.
  for (int it = 0; it < 4; ++it) {
    const int m0 = (it * 8 + wave) << 4;
    const float* arow = gbase + (long)(m0 + (lane & 15)) * CC + half * 8;
    if (it < 3) __builtin_prefetch(arow + 128 * CC, 0, 1);
    v8f acc = {0.f, 0.f, 0.f, 0.f, 0.f, 0.f, 0.f, 0.f};
#pragma unroll
    for (int ks = 0; ks < 8; ++ks) {
      const float* p = arow + ks * 32;
      v16bf af;
#pragma unroll
      for (int i = 0; i < 8; ++i) af[i] = (__bf16)p[i];
#pragma unroll
      for (int i = 0; i < 8; ++i) af[8 + i] = (__bf16)p[16 + i];
      acc = __builtin_amdgcn_wmma_f32_16x16x32_bf16(false, af, false, bfrag[ks],
                                                    (short)0, acc, false, false);
    }
    if (hn < HH) {
      const long sb = ((long)(b * HH + hn) * NN + n) * MM;
#pragma unroll
      for (int rv = 0; rv < 8; ++rv) {
        const int m = m0 + rv + half * 8;
        sc[hn * MM + m] = Sqk[sb + m] + (acc[rv] + gb) * INV_SQRT_D;
      }
    }
  }
  __syncthreads();

  // Softmax over m=512 per head; wave w handles head w.
  {
    const int h = wave;
    float vals[16];
    float mx = -1e30f;
#pragma unroll
    for (int i = 0; i < 16; ++i) {
      vals[i] = sc[h * MM + lane + i * 32];
      mx = fmaxf(mx, vals[i]);
    }
#pragma unroll
    for (int o = 16; o; o >>= 1) mx = fmaxf(mx, __shfl_xor(mx, o, 32));
    float sum = 0.f;
#pragma unroll
    for (int i = 0; i < 16; ++i) {
      vals[i] = __expf(vals[i] - mx);
      sum += vals[i];
    }
#pragma unroll
    for (int o = 16; o; o >>= 1) sum += __shfl_xor(sum, o, 32);
    const float inv = 1.f / sum;
#pragma unroll
    for (int i = 0; i < 16; ++i) sc[h * MM + lane + i * 32] = vals[i] * inv;
  }
  __syncthreads();

  // out[h,d] = sum_m sc[h][m] * V[(m*B+b)*C + h*32+d], + residual.
  const int h = t >> 5, dd = t & 31;
  const float* vb = V + b * CC + h * DD + dd;
  float s = 0.f;
  for (int m = 0; m < MM; ++m) s += sc[h * MM + m] * vb[(long)m * BB * CC];
  const long oidx = ((long)n * BB + b) * CC + h * DD + dd;
  out[oidx] = resid[oidx] + s;
}

// ---------------------------------------------------------------------------
// Row softmax over M=512 for cross-attention scores (in place). Block 128.
// ---------------------------------------------------------------------------
__global__ __launch_bounds__(128) void softmax_rows(float* __restrict__ S) {
  float* p = S + (long)blockIdx.x * MM;
  const int t = threadIdx.x, lane = t & 31, wave = t >> 5;
  __shared__ float red[4];
  float v[4];
  float mx = -1e30f;
#pragma unroll
  for (int i = 0; i < 4; ++i) {
    v[i] = p[t + i * 128];
    mx = fmaxf(mx, v[i]);
  }
#pragma unroll
  for (int o = 16; o; o >>= 1) mx = fmaxf(mx, __shfl_xor(mx, o, 32));
  if (!lane) red[wave] = mx;
  __syncthreads();
  mx = fmaxf(fmaxf(red[0], red[1]), fmaxf(red[2], red[3]));
  __syncthreads();
  float sum = 0.f;
#pragma unroll
  for (int i = 0; i < 4; ++i) {
    v[i] = __expf(v[i] - mx);
    sum += v[i];
  }
#pragma unroll
  for (int o = 16; o; o >>= 1) sum += __shfl_xor(sum, o, 32);
  if (!lane) red[wave] = sum;
  __syncthreads();
  sum = red[0] + red[1] + red[2] + red[3];
  const float inv = 1.f / sum;
#pragma unroll
  for (int i = 0; i < 4; ++i) p[t + i * 128] = v[i] * inv;
}

// ---------------------------------------------------------------------------
// Host-side helpers + launcher
// ---------------------------------------------------------------------------
static void launch_gemm(hipStream_t st, bool BT, const float* A, int lda,
                        long sAh, long sAb, const float* Bm, int ldb, long sBh,
                        long sBb, const float* bias, const float* resid,
                        float* Y, int ldy, long sYh, long sYb, int Mr, int Nc,
                        int Kc, int nbatch, int Hb, float alpha, int relu) {
  const int tiles = (Mr / 16) * (Nc / 16);
  dim3 grid((tiles + 7) / 8, nbatch);
  if (BT)
    gemm_wmma_bf16<true><<<grid, 256, 0, st>>>(A, lda, sAh, sAb, Bm, ldb, sBh,
                                               sBb, bias, resid, Y, ldy, sYh,
                                               sYb, Mr, Nc, Kc, Hb, alpha, relu);
  else
    gemm_wmma_bf16<false><<<grid, 256, 0, st>>>(A, lda, sAh, sAb, Bm, ldb, sBh,
                                                sBb, bias, resid, Y, ldy, sYh,
                                                sYb, Mr, Nc, Kc, Hb, alpha, relu);
}

extern "C" void kernel_launch(void* const* d_in, const int* in_sizes, int n_in,
                              void* d_out, int out_size, void* d_ws,
                              size_t ws_size, hipStream_t stream) {
  const float* src        = (const float*)d_in[0];
  const float* tgt        = (const float*)d_in[1];
  const float* src_pos    = (const float*)d_in[2];
  const float* tgt_pos    = (const float*)d_in[3];
  const float* src_global = (const float*)d_in[4];
  const float* tgt_global = (const float*)d_in[5];
  const float* Wq = (const float*)d_in[6];   const float* bq = (const float*)d_in[7];
  const float* Wk = (const float*)d_in[8];   const float* bk = (const float*)d_in[9];
  const float* Wv = (const float*)d_in[10];  const float* bv = (const float*)d_in[11];
  const float* Wg = (const float*)d_in[12];  const float* bg = (const float*)d_in[13];
  const float* Win  = (const float*)d_in[14]; const float* bin  = (const float*)d_in[15];
  const float* Wout = (const float*)d_in[16]; const float* bout = (const float*)d_in[17];
  const float* w1 = (const float*)d_in[18];  const float* b1 = (const float*)d_in[19];
  const float* w2 = (const float*)d_in[20];  const float* b2 = (const float*)d_in[21];
  const float* n1g = (const float*)d_in[22]; const float* n1b = (const float*)d_in[23];
  const float* n2g = (const float*)d_in[24]; const float* n2b = (const float*)d_in[25];
  const float* n3g = (const float*)d_in[26]; const float* n3b = (const float*)d_in[27];

  float* w = (float*)d_ws;
  float* cur_src = w; w += TOK;
  float* cur_tgt = w; w += TOK;
  float* lnb     = w; w += TOK;
  float* lnpos   = w; w += TOK;
  float* Qb      = w; w += TOK;
  float* Kb      = w; w += TOK;
  float* Vb      = w; w += TOK;
  float* s2c     = w; w += TOK;
  float* t2c     = w; w += TOK;
  float* swb     = w; w += TOK;
  float* twb     = w; w += TOK;
  float* Aq_s    = w; w += TOK;
  float* Ak_s    = w; w += TOK;
  float* Av_s    = w; w += TOK;
  float* Aq_t    = w; w += TOK;
  float* Ak_t    = w; w += TOK;
  float* Av_t    = w; w += TOK;
  float* attnb   = w; w += TOK;
  float* U       = w; w += (long)BB * NN * HH * CC;  // 8 MB
  float* GB      = w; w += BB * NN * HH;
  float* Sqk     = w; w += (long)BB * HH * NN * MM;  // 16 MB
  float* hidden  = w; w += (long)NN * BB * FF;       // 4 MB

  const int ROWS = NN * BB;  // 1024 token rows
  const long NM = (long)NN * MM;

  // ===================== self-attention (RPE), src then tgt ================
  for (int which = 0; which < 2; ++which) {
    const float* X    = which ? tgt : src;
    const float* Xpos = which ? tgt_pos : src_pos;
    const float* Xg   = which ? tgt_global : src_global;
    float* cur        = which ? cur_tgt : cur_src;

    ln_kernel<<<ROWS, 256, 0, stream>>>(X, n1g, n1b, Xpos, lnb, lnpos);
    // q,k from ln+pos, v from ln
    launch_gemm(stream, true, lnpos, CC, 0, 0, Wq, CC, 0, 0, bq, nullptr, Qb,
                CC, 0, 0, ROWS, CC, CC, 1, 1, 1.f, 0);
    launch_gemm(stream, true, lnpos, CC, 0, 0, Wk, CC, 0, 0, bk, nullptr, Kb,
                CC, 0, 0, ROWS, CC, CC, 1, 1, 1.f, 0);
    launch_gemm(stream, true, lnb, CC, 0, 0, Wv, CC, 0, 0, bv, nullptr, Vb,
                CC, 0, 0, ROWS, CC, CC, 1, 1, 1.f, 0);
    // u = q-projected Wg rows (per head block)
    u_kernel<<<ROWS, 256, 0, stream>>>(Qb, Wg, bg, U, GB);
    // Sqk[b,h] = (Q_h @ K_h^T) / sqrt(d), batched over 16 (b,h)
    launch_gemm(stream, true, Qb, BB * CC, DD, CC, Kb, BB * CC, DD, CC,
                nullptr, nullptr, Sqk, MM, NM, (long)HH * NM, NN, MM, DD,
                BB * HH, HH, INV_SQRT_D, 0);
    // stream g_in once: fused g-scores + softmax + AV + residual
    rpe_attn_kernel<<<BB * NN, 256, 0, stream>>>(Xg, U, GB, Sqk, Vb, X, cur);
  }

  // ===================== cross attention (standard MHA) ====================
  ln_kernel<<<ROWS, 256, 0, stream>>>(cur_src, n2g, n2b, src_pos, s2c, swb);
  ln_kernel<<<ROWS, 256, 0, stream>>>(cur_tgt, n2g, n2b, tgt_pos, t2c, twb);
  const float* Wqi = Win;                const float* bqi = bin;
  const float* Wki = Win + CC * CC;      const float* bki = bin + CC;
  const float* Wvi = Win + 2 * CC * CC;  const float* bvi = bin + 2 * CC;
  launch_gemm(stream, true, swb, CC, 0, 0, Wqi, CC, 0, 0, bqi, nullptr, Aq_s,
              CC, 0, 0, ROWS, CC, CC, 1, 1, 1.f, 0);
  launch_gemm(stream, true, swb, CC, 0, 0, Wki, CC, 0, 0, bki, nullptr, Ak_s,
              CC, 0, 0, ROWS, CC, CC, 1, 1, 1.f, 0);
  launch_gemm(stream, true, s2c, CC, 0, 0, Wvi, CC, 0, 0, bvi, nullptr, Av_s,
              CC, 0, 0, ROWS, CC, CC, 1, 1, 1.f, 0);
  launch_gemm(stream, true, twb, CC, 0, 0, Wqi, CC, 0, 0, bqi, nullptr, Aq_t,
              CC, 0, 0, ROWS, CC, CC, 1, 1, 1.f, 0);
  launch_gemm(stream, true, twb, CC, 0, 0, Wki, CC, 0, 0, bki, nullptr, Ak_t,
              CC, 0, 0, ROWS, CC, CC, 1, 1, 1.f, 0);
  launch_gemm(stream, true, t2c, CC, 0, 0, Wvi, CC, 0, 0, bvi, nullptr, Av_t,
              CC, 0, 0, ROWS, CC, CC, 1, 1, 1.f, 0);

  for (int dir = 0; dir < 2; ++dir) {
    const float* Aq = dir ? Aq_t : Aq_s;
    const float* Ak = dir ? Ak_s : Ak_t;
    const float* Av = dir ? Av_s : Av_t;
    float* cur      = dir ? cur_tgt : cur_src;
    // scores
    launch_gemm(stream, true, Aq, BB * CC, DD, CC, Ak, BB * CC, DD, CC,
                nullptr, nullptr, Sqk, MM, NM, (long)HH * NM, NN, MM, DD,
                BB * HH, HH, INV_SQRT_D, 0);
    softmax_rows<<<BB * HH * NN, 128, 0, stream>>>(Sqk);
    // attn = P @ V, scattered back into (N,B,C) layout
    launch_gemm(stream, false, Sqk, MM, NM, (long)HH * NM, Av, BB * CC, DD, CC,
                nullptr, nullptr, attnb, BB * CC, DD, CC, NN, DD, MM, BB * HH,
                HH, 1.f, 0);
    // out-proj + residual (in place on cur)
    launch_gemm(stream, true, attnb, CC, 0, 0, Wout, CC, 0, 0, bout, cur, cur,
                CC, 0, 0, ROWS, CC, CC, 1, 1, 1.f, 0);
  }

  // ===================== FFN (shared weights), writes d_out ================
  float* out_f = (float*)d_out;
  for (int which = 0; which < 2; ++which) {
    float* cur = which ? cur_tgt : cur_src;
    float* dst = out_f + (long)which * TOK;
    ln_kernel<<<ROWS, 256, 0, stream>>>(cur, n3g, n3b, nullptr, lnb, nullptr);
    launch_gemm(stream, true, lnb, CC, 0, 0, w1, CC, 0, 0, b1, nullptr, hidden,
                FF, 0, 0, ROWS, FF, CC, 1, 1, 1.f, 1);
    launch_gemm(stream, true, hidden, FF, 0, 0, w2, FF, 0, 0, b2, cur, dst,
                CC, 0, 0, ROWS, CC, FF, 1, 1, 1.f, 0);
  }
}
